// FFJORD_67465346285802
// MI455X (gfx1250) — matrix operand order repaired and checked by reading, not compiled, not run on hardware
//
#include <hip/hip_runtime.h>

// ---------------------------------------------------------------------------
// FFJORD (2 bijectors x 16 DOPRI5 steps x 6 MLP stages) as a single persistent
// CDNA5 kernel. Mixed precision: f16 WMMA (v_wmma_f32_16x16x32_f16) with f32
// accumulation; RK45 state in f32 registers; weights resident in LDS.
// Round 2: hoist shared A-fragments into registers, interleave two N-tiles
// per inner loop (independent WMMA chains -> fewer hazard NOPs / dscnt stalls).
// ---------------------------------------------------------------------------

typedef _Float16 v16h __attribute__((ext_vector_type(16)));
typedef _Float16 v8h  __attribute__((ext_vector_type(8)));
typedef float    v8f  __attribute__((ext_vector_type(8)));

#define D_DIM    64
#define H_DIM    256
#define B_DIM    4096
#define NSTEPS   16
#define M_TILE   32          // batch rows per workgroup -> 128 workgroups
#define NTHREADS 256         // 8 wave32s

// LDS row strides (in halves) padded to de-skew banks; multiples of 8 halves
// keep every 16B fragment load aligned.
#define W1_STRIDE 72         // 64  + 8
#define W2_STRIDE 264        // 256 + 8
#define A_STRIDE  72
#define H_STRIDE  264

// Dynamic-LDS layout (bytes); every offset is 16B aligned.
#define OFF_W2T 0
#define OFF_W1T (OFF_W2T + H_DIM * W2_STRIDE * 2)    // 135168
#define OFF_W3T (OFF_W1T + H_DIM * W1_STRIDE * 2)    // +36864
#define OFF_H   (OFF_W3T + D_DIM * W2_STRIDE * 2)    // +33792
#define OFF_H2  (OFF_H   + M_TILE * H_STRIDE * 2)    // +16896
#define OFF_A   (OFF_H2  + M_TILE * H_STRIDE * 2)    // +16896
#define OFF_B1  (OFF_A   + M_TILE * A_STRIDE * 2)    // +4608
#define OFF_B2  (OFF_B1  + H_DIM * 4)
#define OFF_B3  (OFF_B2  + H_DIM * 4)
#define OFF_W1L (OFF_B3  + 256)                      // b3: 64 floats
#define SMEM_BYTES (OFF_W1L + H_DIM * 4)             // ~241.8 KB < 320 KB/WGP

#define V8F_ZERO {0.f, 0.f, 0.f, 0.f, 0.f, 0.f, 0.f, 0.f}

static __device__ __forceinline__ v16h cat16(v8h lo, v8h hi) {
  return __builtin_shufflevector(lo, hi, 0, 1, 2, 3, 4, 5, 6, 7,
                                 8, 9, 10, 11, 12, 13, 14, 15);
}

// tanh(x) = 1 - 2/(exp2(2x*log2e)+1)  -> v_exp_f32 + v_rcp_f32
static __device__ __forceinline__ float fast_tanh(float x) {
  float e = __builtin_amdgcn_exp2f(x * 2.8853900817779268f);
  float r = __builtin_amdgcn_rcpf(e + 1.0f);
  return 1.0f - 2.0f * r;
}

// A fragment (16x32 f16, ISA 7.12.2): lane m=l&15; K chunks at
// kk*32 + (l>>4)*8 and +16 (two ds_load_b128).
static __device__ __forceinline__ v16h load_afrag(const _Float16* base,
                                                  int rowBase, int stride,
                                                  int kk, int l15, int lhi) {
  const _Float16* p = base + (rowBase + l15) * stride + kk * 32 + lhi * 8;
  v8h lo = *(const v8h*)p;
  v8h hi = *(const v8h*)(p + 16);
  return cat16(lo, hi);
}

// B fragment (32x16 f16): lane n=l&15 of W^T row n; K chunks at
// kk*32 + (l>>4)*8 and +16 (two ds_load_b128).
static __device__ __forceinline__ v16h load_bfrag(const _Float16* wt,
                                                  int nt, int stride,
                                                  int kk, int l15, int lhi) {
  const _Float16* p = wt + (nt * 16 + l15) * stride + kk * 32 + lhi * 8;
  v8h lo = *(const v8h*)p;
  v8h hi = *(const v8h*)(p + 16);
  return cat16(lo, hi);
}

// bias + tanh epilogue, f16 store in D-matrix layout.
static __device__ __forceinline__ void epi_tanh(v8f acc, float bias,
                                                _Float16* dst, int rowBase,
                                                int stride, int col, int lhi) {
#pragma unroll
  for (int r = 0; r < 8; ++r) {
    float hv = fast_tanh(acc[r] + bias);
    dst[(rowBase + r + 8 * lhi) * stride + col] = (_Float16)hv;
  }
}

extern "C" __global__ void __launch_bounds__(NTHREADS, 1)
ffjord_kernel(const float* __restrict__ x, float* __restrict__ out,
              const float* __restrict__ W1a, const float* __restrict__ b1a,
              const float* __restrict__ W2a, const float* __restrict__ b2a,
              const float* __restrict__ W3a, const float* __restrict__ b3a,
              const float* __restrict__ W1b, const float* __restrict__ b1b,
              const float* __restrict__ W2b, const float* __restrict__ b2b,
              const float* __restrict__ W3b, const float* __restrict__ b3b) {
  extern __shared__ __align__(16) char smem[];
  _Float16* sW2t = (_Float16*)(smem + OFF_W2T);
  _Float16* sW1t = (_Float16*)(smem + OFF_W1T);
  _Float16* sW3t = (_Float16*)(smem + OFF_W3T);
  _Float16* sH   = (_Float16*)(smem + OFF_H);
  _Float16* sH2  = (_Float16*)(smem + OFF_H2);
  _Float16* sA   = (_Float16*)(smem + OFF_A);
  float* sb1  = (float*)(smem + OFF_B1);
  float* sb2  = (float*)(smem + OFF_B2);
  float* sb3  = (float*)(smem + OFF_B3);
  float* sw1l = (float*)(smem + OFF_W1L);

  const int tid  = threadIdx.x;
  const int lane = tid & 31;
  const int wv   = tid >> 5;
  const int l15  = lane & 15;
  const int lhi  = lane >> 4;          // 0 or 1
  const int rBase = (wv >> 2) * 16;    // this wave's 16 batch rows (local)
  const int cGrp  = wv & 3;            // this wave's 16-col group of D
  const int rowBlock = blockIdx.x * M_TILE;

  // Register patch of the ODE state: D-matrix layout (row = rBase+r+8*lhi,
  // col = cGrp*16 + l15), so WMMA results land directly on it.
  float y[8];
#pragma unroll
  for (int r = 0; r < 8; ++r)
    y[r] = x[(rowBlock + rBase + r + 8 * lhi) * D_DIM + cGrp * 16 + l15];

  const float hstep = 1.0f / NSTEPS;

  // Dormand-Prince tableau.
  constexpr float AT[6][5] = {
      {0.f, 0.f, 0.f, 0.f, 0.f},
      {1.f / 5.f, 0.f, 0.f, 0.f, 0.f},
      {3.f / 40.f, 9.f / 40.f, 0.f, 0.f, 0.f},
      {44.f / 45.f, -56.f / 15.f, 32.f / 9.f, 0.f, 0.f},
      {19372.f / 6561.f, -25360.f / 2187.f, 64448.f / 6561.f, -212.f / 729.f, 0.f},
      {9017.f / 3168.f, -355.f / 33.f, 46732.f / 5247.f, 49.f / 176.f, -5103.f / 18656.f}};
  constexpr float CC[6] = {0.f, 0.2f, 0.3f, 0.8f, 8.f / 9.f, 1.f};
  constexpr float BW[6] = {35.f / 384.f, 0.f, 500.f / 1113.f, 125.f / 192.f,
                           -2187.f / 6784.f, 11.f / 84.f};

#pragma unroll 1
  for (int bj = 0; bj < 2; ++bj) {
    const float* W1 = bj ? W1b : W1a;
    const float* B1 = bj ? b1b : b1a;
    const float* W2 = bj ? W2b : W2a;
    const float* B2 = bj ? b2b : b2a;
    const float* W3 = bj ? W3b : W3a;
    const float* B3 = bj ? b3b : b3a;

    __syncthreads();  // previous bijector done reading weight LDS
    {
      // Cooperative f32 -> f16 transposed weight stage-in. Per fixed k, the
      // 256 threads read 256 contiguous floats (fully coalesced, L2-resident).
      const int n = tid;
      for (int k = 0; k < D_DIM; ++k)
        sW1t[n * W1_STRIDE + k] = (_Float16)W1[k * H_DIM + n];
      sw1l[n] = W1[D_DIM * H_DIM + n];  // the t-row of W1 -> time bias
      sb1[n] = B1[n];
      sb2[n] = B2[n];
      if (n < D_DIM) sb3[n] = B3[n];
      for (int k = 0; k < H_DIM; ++k)
        sW2t[n * W2_STRIDE + k] = (_Float16)W2[k * H_DIM + n];
      if (n < D_DIM)
        for (int k = 0; k < H_DIM; ++k)
          sW3t[n * W2_STRIDE + k] = (_Float16)W3[k * D_DIM + n];
    }
    __syncthreads();

#pragma unroll 1
    for (int step = 0; step < NSTEPS; ++step) {
      const float t0 = (float)step * hstep;
      float kst[6][8];

#pragma unroll
      for (int s = 0; s < 6; ++s) {
        // ---- stage state y_s = y + h * sum_j a[s][j] k_j  -> aBuf (f16)
#pragma unroll
        for (int r = 0; r < 8; ++r) {
          float a = y[r];
#pragma unroll
          for (int j = 0; j < 5; ++j)
            if (j < s) a += hstep * AT[s][j] * kst[j][r];
          sA[(rBase + r + 8 * lhi) * A_STRIDE + cGrp * 16 + l15] = (_Float16)a;
        }
        __syncthreads();
        const float ts = t0 + hstep * CC[s];

        // ---- layer 1: [32x64] @ [64x256], K=64 (2 WMMA k-steps), bias+t, tanh
        {
          v16h a0 = load_afrag(sA, rBase, A_STRIDE, 0, l15, lhi);
          v16h a1 = load_afrag(sA, rBase, A_STRIDE, 1, l15, lhi);
#pragma unroll
          for (int j = 0; j < 4; j += 2) {
            const int nt0 = cGrp * 4 + j, nt1 = nt0 + 1;
            v8f acc0 = V8F_ZERO, acc1 = V8F_ZERO;
            v16h b00 = load_bfrag(sW1t, nt0, W1_STRIDE, 0, l15, lhi);
            v16h b10 = load_bfrag(sW1t, nt1, W1_STRIDE, 0, l15, lhi);
            v16h b01 = load_bfrag(sW1t, nt0, W1_STRIDE, 1, l15, lhi);
            v16h b11 = load_bfrag(sW1t, nt1, W1_STRIDE, 1, l15, lhi);
            acc0 = __builtin_amdgcn_wmma_f32_16x16x32_f16(
                false, a0, false, b00, (short)0, acc0, false, false);
            acc1 = __builtin_amdgcn_wmma_f32_16x16x32_f16(
                false, a0, false, b10, (short)0, acc1, false, false);
            acc0 = __builtin_amdgcn_wmma_f32_16x16x32_f16(
                false, a1, false, b01, (short)0, acc0, false, false);
            acc1 = __builtin_amdgcn_wmma_f32_16x16x32_f16(
                false, a1, false, b11, (short)0, acc1, false, false);
            const int c0 = nt0 * 16 + l15, c1 = nt1 * 16 + l15;
            epi_tanh(acc0, sb1[c0] + ts * sw1l[c0], sH, rBase, H_STRIDE, c0, lhi);
            epi_tanh(acc1, sb1[c1] + ts * sw1l[c1], sH, rBase, H_STRIDE, c1, lhi);
          }
        }
        __syncthreads();

        // ---- layer 2: [32x256] @ [256x256], K=256 (8 WMMA k-steps), tanh
        {
          v16h a2[8];
#pragma unroll
          for (int kk = 0; kk < 8; ++kk)
            a2[kk] = load_afrag(sH, rBase, H_STRIDE, kk, l15, lhi);
#pragma unroll
          for (int j = 0; j < 4; j += 2) {
            const int nt0 = cGrp * 4 + j, nt1 = nt0 + 1;
            v8f acc0 = V8F_ZERO, acc1 = V8F_ZERO;
#pragma unroll
            for (int kk = 0; kk < 8; ++kk) {
              v16h b0 = load_bfrag(sW2t, nt0, W2_STRIDE, kk, l15, lhi);
              v16h b1 = load_bfrag(sW2t, nt1, W2_STRIDE, kk, l15, lhi);
              acc0 = __builtin_amdgcn_wmma_f32_16x16x32_f16(
                  false, a2[kk], false, b0, (short)0, acc0, false, false);
              acc1 = __builtin_amdgcn_wmma_f32_16x16x32_f16(
                  false, a2[kk], false, b1, (short)0, acc1, false, false);
            }
            const int c0 = nt0 * 16 + l15, c1 = nt1 * 16 + l15;
            epi_tanh(acc0, sb2[c0], sH2, rBase, H_STRIDE, c0, lhi);
            epi_tanh(acc1, sb2[c1], sH2, rBase, H_STRIDE, c1, lhi);
          }
        }
        __syncthreads();

        // ---- layer 3: [32x256] @ [256x64]; each wave's single 16x16 tile is
        // exactly its register patch. Two K-split accumulators break the
        // serial WMMA accumulation chain.
        {
          v8f acc0 = V8F_ZERO, acc1 = V8F_ZERO;
#pragma unroll
          for (int kk = 0; kk < 8; kk += 2) {
            v16h ae = load_afrag(sH2, rBase, H_STRIDE, kk, l15, lhi);
            v16h ao = load_afrag(sH2, rBase, H_STRIDE, kk + 1, l15, lhi);
            v16h be = load_bfrag(sW3t, cGrp, W2_STRIDE, kk, l15, lhi);
            v16h bo = load_bfrag(sW3t, cGrp, W2_STRIDE, kk + 1, l15, lhi);
            acc0 = __builtin_amdgcn_wmma_f32_16x16x32_f16(
                false, ae, false, be, (short)0, acc0, false, false);
            acc1 = __builtin_amdgcn_wmma_f32_16x16x32_f16(
                false, ao, false, bo, (short)0, acc1, false, false);
          }
          const float bias = sb3[cGrp * 16 + l15];
#pragma unroll
          for (int r = 0; r < 8; ++r) kst[s][r] = acc0[r] + acc1[r] + bias;
        }
      }  // stages

      // ---- 5th-order update (BW[1] == 0 folds away)
#pragma unroll
      for (int r = 0; r < 8; ++r) {
        float d = BW[0] * kst[0][r] + BW[2] * kst[2][r] + BW[3] * kst[3][r] +
                  BW[4] * kst[4][r] + BW[5] * kst[5][r];
        y[r] += hstep * d;
      }
    }  // steps
  }    // bijectors

#pragma unroll
  for (int r = 0; r < 8; ++r)
    out[(rowBlock + rBase + r + 8 * lhi) * D_DIM + cGrp * 16 + l15] = y[r];
}

extern "C" void kernel_launch(void* const* d_in, const int* in_sizes, int n_in,
                              void* d_out, int out_size, void* d_ws,
                              size_t ws_size, hipStream_t stream) {
  (void)in_sizes; (void)n_in; (void)d_ws; (void)ws_size; (void)out_size;
  const float* x = (const float*)d_in[0];
  const float* p[12];
  for (int i = 0; i < 12; ++i) p[i] = (const float*)d_in[1 + i];
  dim3 grid(B_DIM / M_TILE);   // 128 workgroups (one per WGP; ~242 KB LDS each)
  dim3 block(NTHREADS);        // 8 wave32s
  ffjord_kernel<<<grid, block, SMEM_BYTES, stream>>>(
      x, (float*)d_out, p[0], p[1], p[2], p[3], p[4], p[5], p[6], p[7], p[8],
      p[9], p[10], p[11]);
}
